// SelectiveLinear_62740882260718
// MI455X (gfx1250) — matrix operand order repaired and checked by reading, not compiled, not run on hardware
//
#include <hip/hip_runtime.h>

#define BATCH 1024
#define IN_F  4096
#define OUT_F 4096
#define TOPK  64
#define LDS_STR 40   // bf16 elems per LDS row (80B = 20 banks): conflict-free b128 reads

typedef __attribute__((ext_vector_type(16))) __bf16       v16bf;
typedef __attribute__((ext_vector_type(8)))  float        v8f;
typedef __attribute__((ext_vector_type(4)))  unsigned int v4u;
typedef __attribute__((ext_vector_type(2)))  unsigned int v2u;
typedef __attribute__((ext_vector_type(4)))  int          v4i;

union FragBF { v16bf bf; v4u q[2]; };

#if __has_builtin(__builtin_amdgcn_global_load_async_to_lds_b128)
#define SL_ASYNC 1
#else
#define SL_ASYNC 0
#endif

__device__ __forceinline__ unsigned short f2bf(float f) {
  unsigned int u = __float_as_uint(f);
  u = u + 0x7FFFu + ((u >> 16) & 1u);          // round-to-nearest-even
  return (unsigned short)(u >> 16);
}
__device__ __forceinline__ float bf2f(unsigned short h) {
  return __uint_as_float(((unsigned int)h) << 16);
}

// 16B global -> LDS stage; async (no VGPR round-trip, ASYNCcnt-tracked) if available
__device__ __forceinline__ void ld16_to_lds(const unsigned short* g, unsigned short* l) {
#if SL_ASYNC
  __builtin_amdgcn_global_load_async_to_lds_b128(
      (__attribute__((address_space(1))) v4i*)(v4i*)(void*)g,
      (__attribute__((address_space(3))) v4i*)(v4i*)(void*)l,
      0, 0);
#else
  *(v4u*)l = *(const v4u*)g;
#endif
}

__device__ __forceinline__ void wait_stage() {
#if SL_ASYNC
#if __has_builtin(__builtin_amdgcn_s_wait_asynccnt)
  __builtin_amdgcn_s_wait_asynccnt(0);
#else
  asm volatile("s_wait_asynccnt 0x0" ::: "memory");
#endif
#endif
}

// ---------------------------------------------------------------------------
// Kernel 0: one-shot weight split fp32 -> bf16 hi/lo planes (streamed once).
// ---------------------------------------------------------------------------
__global__ __launch_bounds__(256)
void sl_wsplit_k(const float* __restrict__ W,
                 unsigned short* __restrict__ Whi, unsigned short* __restrict__ Wlo) {
  const size_t i = ((size_t)blockIdx.x * 256 + threadIdx.x) * 4;
  float f[4];
  *(v4u*)f = *(const v4u*)(W + i);
  unsigned int hw[2], lw[2];
#pragma unroll
  for (int j = 0; j < 2; ++j) {
    unsigned short h0 = f2bf(f[2 * j]),     h1 = f2bf(f[2 * j + 1]);
    unsigned short l0 = f2bf(f[2 * j]     - bf2f(h0));
    unsigned short l1 = f2bf(f[2 * j + 1] - bf2f(h1));
    hw[j] = (unsigned int)h0 | ((unsigned int)h1 << 16);
    lw[j] = (unsigned int)l0 | ((unsigned int)l1 << 16);
  }
  v2u hv = { hw[0], hw[1] };
  v2u lv = { lw[0], lw[1] };
  *(v2u*)(Whi + i) = hv;
  *(v2u*)(Wlo + i) = lv;
}

// ---------------------------------------------------------------------------
// Kernel 1: scatter sparse (x, idx) into dense row, split into bf16 hi/lo.
// ---------------------------------------------------------------------------
__global__ __launch_bounds__(128)
void sl_scatter_k(const float* __restrict__ x, const int* __restrict__ idx,
                  unsigned short* __restrict__ Xhi, unsigned short* __restrict__ Xlo) {
  __shared__ float dense[IN_F];                 // 16 KB
  const int b = blockIdx.x, t = threadIdx.x;
  for (int c = t; c < IN_F; c += 128) dense[c] = 0.0f;
  __syncthreads();
  if (t < TOPK) {
    atomicAdd(&dense[idx[b * TOPK + t]], x[b * TOPK + t]);   // ds_add_f32, dup-safe
  }
  __syncthreads();
  for (int c = t; c < IN_F; c += 128) {
    float v = dense[c];
    unsigned short h = f2bf(v);
    unsigned short l = f2bf(v - bf2f(h));
    Xhi[(size_t)b * IN_F + c] = h;
    Xlo[(size_t)b * IN_F + c] = l;
  }
}

// ---------------------------------------------------------------------------
// Kernel 2: R = X(1024x4096) * W^T + bias via v_wmma_f32_16x16x32_bf16,
// bf16 hi/lo 3-product split, double-buffered async global->LDS staging.
// Block tile 64(M) x 64(N), K-step 32, 8 waves, each wave -> 16x32 of D.
// ---------------------------------------------------------------------------
__global__ __launch_bounds__(256)
void sl_gemm_k(const unsigned short* __restrict__ Xhi,
               const unsigned short* __restrict__ Xlo,
               const unsigned short* __restrict__ Whi,
               const unsigned short* __restrict__ Wlo,
               const float* __restrict__ bias,
               float* __restrict__ R) {
  // [buffer][plane: Ahi, Alo, Bhi, Blo][64 x LDS_STR]  -> 2*4*5120B = 40 KB
  __shared__ unsigned short sm[2][4][64 * LDS_STR];

  const int t    = threadIdx.x;
  const int lane = t & 31;                 // wave32
  const int wave = t >> 5;                 // 8 waves
  const int wm   = wave & 3;               // 4 M sub-tiles of 16
  const int wn   = wave >> 2;              // 2 N groups of 32
  const int bm0  = blockIdx.y * 64;        // batch rows
  const int bn0  = blockIdx.x * 64;        // output cols

  const int lrow = t >> 2;                 // 0..63 : tile-stage row
  const int lcol = (t & 3) * 8;            // 0,8,16,24
  const int loff = lrow * LDS_STR + lcol;
  const size_t gA = (size_t)(bm0 + lrow) * IN_F + lcol;
  const size_t gB = (size_t)(bn0 + lrow) * IN_F + lcol;

  v8f acc0 = {}; v8f acc1 = {};

#define SL_STAGE(bufi, k0)                                   \
  do {                                                       \
    ld16_to_lds(Xhi + gA + (k0), &sm[bufi][0][loff]);        \
    ld16_to_lds(Xlo + gA + (k0), &sm[bufi][1][loff]);        \
    ld16_to_lds(Whi + gB + (k0), &sm[bufi][2][loff]);        \
    ld16_to_lds(Wlo + gB + (k0), &sm[bufi][3][loff]);        \
  } while (0)

  SL_STAGE(0, 0);

  for (int k0 = 0; k0 < IN_F; k0 += 32) {
    const int cur = (k0 >> 5) & 1;
    wait_stage();
    __syncthreads();                       // staged tile visible to all waves
    if (k0 + 32 < IN_F) SL_STAGE(cur ^ 1, k0 + 32);

    // A fragments (16x32 bf16): lane<16 -> row, K 0-15; lane>=16 -> K 16-31
    FragBF ah, al;
    {
      const int r  = wm * 16 + (lane & 15);
      const int c0 = (lane >> 4) * 16;
      ah.q[0] = *(const v4u*)&sm[cur][0][r * LDS_STR + c0];
      ah.q[1] = *(const v4u*)&sm[cur][0][r * LDS_STR + c0 + 8];
      al.q[0] = *(const v4u*)&sm[cur][1][r * LDS_STR + c0];
      al.q[1] = *(const v4u*)&sm[cur][1][r * LDS_STR + c0 + 8];
    }
    // B fragments (32x16 bf16, two N sub-tiles): lane<16 -> col n, K 0-15; lane>=16 -> K 16-31
    FragBF bh0, bl0, bh1, bl1;
    {
      const int kc = (lane >> 4) * 16;
      const int c0 = wn * 32 + (lane & 15);
      const int c1 = c0 + 16;
      bh0.q[0] = *(const v4u*)&sm[cur][2][c0 * LDS_STR + kc];
      bh0.q[1] = *(const v4u*)&sm[cur][2][c0 * LDS_STR + kc + 8];
      bl0.q[0] = *(const v4u*)&sm[cur][3][c0 * LDS_STR + kc];
      bl0.q[1] = *(const v4u*)&sm[cur][3][c0 * LDS_STR + kc + 8];
      bh1.q[0] = *(const v4u*)&sm[cur][2][c1 * LDS_STR + kc];
      bh1.q[1] = *(const v4u*)&sm[cur][2][c1 * LDS_STR + kc + 8];
      bl1.q[0] = *(const v4u*)&sm[cur][3][c1 * LDS_STR + kc];
      bl1.q[1] = *(const v4u*)&sm[cur][3][c1 * LDS_STR + kc + 8];
    }

    // hi*hi + hi*lo + lo*hi (drop lo*lo): ~fp32-accurate product
    acc0 = __builtin_amdgcn_wmma_f32_16x16x32_bf16(false, ah.bf, false, bh0.bf, (short)0, acc0, false, false);
    acc0 = __builtin_amdgcn_wmma_f32_16x16x32_bf16(false, ah.bf, false, bl0.bf, (short)0, acc0, false, false);
    acc0 = __builtin_amdgcn_wmma_f32_16x16x32_bf16(false, al.bf, false, bh0.bf, (short)0, acc0, false, false);
    acc1 = __builtin_amdgcn_wmma_f32_16x16x32_bf16(false, ah.bf, false, bh1.bf, (short)0, acc1, false, false);
    acc1 = __builtin_amdgcn_wmma_f32_16x16x32_bf16(false, ah.bf, false, bl1.bf, (short)0, acc1, false, false);
    acc1 = __builtin_amdgcn_wmma_f32_16x16x32_bf16(false, al.bf, false, bh1.bf, (short)0, acc1, false, false);
  }
#undef SL_STAGE

  // epilogue: D layout lanes 0-15 -> M 0-7 (vgpr idx), lanes 16-31 -> M 8-15
  const int n    = lane & 15;
  const int half = lane >> 4;
  const int col0 = bn0 + wn * 32 + n;
  const int col1 = col0 + 16;
  const float b0 = bias[col0], b1 = bias[col1];
  const int rowb = bm0 + wm * 16 + half * 8;
#pragma unroll
  for (int v = 0; v < 8; ++v) {
    R[(size_t)(rowb + v) * OUT_F + col0] = acc0[v] + b0;
    R[(size_t)(rowb + v) * OUT_F + col1] = acc1[v] + b1;
  }
}

// ---------------------------------------------------------------------------
// Kernel 3: per-row top-64 (descending, smallest index on ties), one block/row.
// ---------------------------------------------------------------------------
__global__ __launch_bounds__(256)
void sl_topk_k(const float* __restrict__ R, float* __restrict__ outv, int* __restrict__ outi) {
  __shared__ float vals[OUT_F];                       // 16 KB
  __shared__ unsigned long long red[256];
  const int b = blockIdx.x, t = threadIdx.x;
  for (int j = t; j < OUT_F; j += 256) vals[j] = R[(size_t)b * OUT_F + j];
  __syncthreads();

  for (int it = 0; it < TOPK; ++it) {
    unsigned long long best = 0ull;
    for (int j = t; j < OUT_F; j += 256) {
      unsigned int u = __float_as_uint(vals[j]);
      u ^= (u >> 31) ? 0xFFFFFFFFu : 0x80000000u;     // monotonic float key
      unsigned long long p = ((unsigned long long)u << 32) |
                             (unsigned long long)(unsigned int)(OUT_F - 1 - j);
      if (p > best) best = p;
    }
    red[t] = best;
    __syncthreads();
#pragma unroll
    for (int s = 128; s > 0; s >>= 1) {
      if (t < s) { if (red[t + s] > red[t]) red[t] = red[t + s]; }
      __syncthreads();
    }
    const unsigned long long w = red[0];
    const int j = (OUT_F - 1) - (int)(w & 0xFFFFFFFFull);
    if (t == 0) {
      unsigned int u = (unsigned int)(w >> 32);
      u = (u >> 31) ? (u ^ 0x80000000u) : ~u;
      outv[b * TOPK + it] = __uint_as_float(u);
      outi[b * TOPK + it] = j;
      vals[j] = -__builtin_inff();
    }
    __syncthreads();
  }
}

// ---------------------------------------------------------------------------
extern "C" void kernel_launch(void* const* d_in, const int* in_sizes, int n_in,
                              void* d_out, int out_size, void* d_ws, size_t ws_size,
                              hipStream_t stream) {
  const float* input  = (const float*)d_in[0];   // (B, 64)
  const float* weight = (const float*)d_in[1];   // (O, I) row-major
  const float* bias   = (const float*)d_in[2];   // (O,)
  const int*   idx    = (const int*)d_in[3];     // (B, 64)

  // workspace: Xhi 8MB | Xlo 8MB | Whi 32MB | Wlo 32MB | R 16MB  -> 96 MB
  unsigned short* Xhi = (unsigned short*)d_ws;
  unsigned short* Xlo = Xhi + (size_t)BATCH * IN_F;
  unsigned short* Whi = Xlo + (size_t)BATCH * IN_F;
  unsigned short* Wlo = Whi + (size_t)OUT_F * IN_F;
  float*          R   = (float*)(Wlo + (size_t)OUT_F * IN_F);

  sl_wsplit_k<<<(OUT_F * (size_t)IN_F) / (256 * 4), 256, 0, stream>>>(weight, Whi, Wlo);
  sl_scatter_k<<<BATCH, 128, 0, stream>>>(input, idx, Xhi, Xlo);

  dim3 grid(OUT_F / 64, BATCH / 64);
  sl_gemm_k<<<grid, 256, 0, stream>>>(Xhi, Xlo, Whi, Wlo, bias, R);

  float* outv = (float*)d_out;                       // values first
  int*   outi = (int*)d_out + (size_t)BATCH * TOPK;  // then indices
  sl_topk_k<<<BATCH, 256, 0, stream>>>(R, outv, outi);
}
